// QKVAttention_38474317037920
// MI455X (gfx1250) — matrix operand order repaired and checked by reading, not compiled, not run on hardware
//
#include <hip/hip_runtime.h>

// ---------- fixed problem sizes ----------
#define BSZ   4
#define TSEQ  2048
#define CDIM  1024
#define NHEAD 16
#define DHEAD 64
#define MROWS (BSZ * TSEQ)   // 8192

typedef __bf16 bf16;
typedef __attribute__((ext_vector_type(16))) __bf16 bf16x16;
typedef __attribute__((ext_vector_type(8)))  __bf16 bf16x8;
typedef __attribute__((ext_vector_type(4)))  __bf16 bf16x4;
typedef __attribute__((ext_vector_type(8)))  float  v8f;
typedef __attribute__((ext_vector_type(4)))  unsigned u32x4;
typedef __attribute__((ext_vector_type(8)))  int      i32x8;
typedef __attribute__((ext_vector_type(4)))  int      i32x4;

__device__ __forceinline__ bf16 f2bf(float f) {
    union { float f; unsigned u; } in; in.f = f;
    unsigned u = in.u + 0x7fffu + ((in.u >> 16) & 1u);   // RNE
    union { unsigned short s; bf16 b; } out;
    out.s = (unsigned short)(u >> 16);
    return out.b;
}

__device__ __forceinline__ bf16x16 cat8(bf16x8 lo, bf16x8 hi) {
    return __builtin_shufflevector(lo, hi, 0,1,2,3,4,5,6,7,8,9,10,11,12,13,14,15);
}

// A fragment: 16x32 bf16, row-major source [m][k] (ld = k-stride).
// Lane holds row m=lane%16; elems j<8 -> K=8*(lane/16)+j, j>=8 -> K=16+8*(lane/16)+(j-8).
__device__ __forceinline__ bf16x16 load_a_frag(const bf16* p, int ld) {
    int lane = threadIdx.x & 31;
    const bf16* r = p + (size_t)(lane & 15) * ld + ((lane >> 4) << 3);
    bf16x8 lo = *(const bf16x8*)r;
    bf16x8 hi = *(const bf16x8*)(r + 16);
    return cat8(lo, hi);
}

// B fragment: logical 32x16 B[k][n] from TRANSPOSED storage [n][k] (ld = k-stride).
// Lane holds col n=lane%16; elems j<8 -> K=8*(lane/16)+j, j>=8 -> K=16+8*(lane/16)+(j-8).
__device__ __forceinline__ bf16x16 load_bt_frag(const bf16* p, int ld) {
    int lane = threadIdx.x & 31;
    const bf16* r = p + (size_t)(lane & 15) * ld + ((lane >> 4) << 3);
    bf16x8 lo = *(const bf16x8*)r;
    bf16x8 hi = *(const bf16x8*)(r + 16);
    return cat8(lo, hi);
}

__device__ __forceinline__ v8f wmma_bf16(bf16x16 a, bf16x16 b, v8f c) {
    return __builtin_amdgcn_wmma_f32_16x16x32_bf16(false, a, false, b,
                                                   (short)0, c, false, false);
}

// ----------------------------------------------------------------------
// TDM: DMA a 128-row x 32-col bf16 tile (row stride 1024 elem) to LDS.
// D# per CDNA5 ISA §8.3/8.4: group0 = {count/type, lds_addr, global_addr},
// group1 = {data_size, tensor dims, tile dims, stride}. 2D -> groups 2/3 zero.
// This toolchain's builtin takes 6 args (extra i32x8 group, then cpol).
// ----------------------------------------------------------------------
__device__ __forceinline__ void tdm_load_tile(unsigned lds_addr,
                                              unsigned long long ga) {
    u32x4 g0;
    g0[0] = 1u;                                             // count=1, user mode
    g0[1] = lds_addr;                                       // LDS byte address
    g0[2] = (unsigned)ga;                                   // global_addr[31:0]
    g0[3] = (unsigned)((ga >> 32) & 0x01FFFFFFu) | (2u << 30); // ga[56:32] | type=2
    i32x8 g1;
    g1[0] = (int)(1u << 16);          // data_size code 1 = 2 bytes
    g1[1] = 0;                        // atomic barrier off; tensor_dim0[15:0]=0
    g1[2] = 0x10;                     // tensor_dim0 = 1<<20 (no OOB), dim1 lo = 0
    g1[3] = (int)(0x10u | (32u << 16));   // tensor_dim1 = 1<<20 ; tile_dim0 = 32
    g1[4] = 128;                      // tile_dim1 = 128, tile_dim2 = 0
    g1[5] = 1024;                     // tensor_dim0_stride = 1024 elements
    g1[6] = 0;
    g1[7] = 0;
    i32x4 z4 = (i32x4)(0);
    i32x8 z8 = (i32x8)(0);
    __builtin_amdgcn_tensor_load_to_lds(g0, g1, z4, z4, z8, 0);
}

// ======================================================================
// GEMM: out[8192][1024] = alpha * A(bf16,[m][1024]) * B(bf16,[n][1024]^T)
// STORE: 0 = f32 row-major [m][n] (final output)
//        1 = bf16 head-split  [bh][t][d]   (q, k)
//        2 = bf16 head-split T [bh][d][t]  (v)
// Tiles staged by the Tensor Data Mover, double-buffered in LDS.
// ======================================================================
template <int STORE>
__global__ __launch_bounds__(256) void gemm_tdm_wmma(
    const bf16* __restrict__ A, const bf16* __restrict__ Bm,
    void* __restrict__ out, float alpha)
{
    constexpr int KD = CDIM, ND = CDIM;
    __shared__ bf16 lA[2][128][32];
    __shared__ bf16 lB[2][128][32];

    const int tid  = threadIdx.x;
    const int wave = tid >> 5;
    const int lane = tid & 31;
    const int hv   = lane >> 4;
    const int ln   = lane & 15;
    const int n0   = blockIdx.x * 128;
    const int m0   = blockIdx.y * 128;

    const unsigned long long gaA =
        (unsigned long long)(uintptr_t)(A + (size_t)m0 * KD);
    const unsigned long long gaB =
        (unsigned long long)(uintptr_t)(Bm + (size_t)n0 * KD);

    v8f acc[8];
#pragma unroll
    for (int j = 0; j < 8; ++j) acc[j] = (v8f)(0.0f);

    // prologue: DMA tile 0 into buffer 0 (wave 0 issues; TDM ignores EXEC)
    if (wave == 0) {
        tdm_load_tile((unsigned)(uintptr_t)&lA[0][0][0], gaA);
        tdm_load_tile((unsigned)(uintptr_t)&lB[0][0][0], gaB);
    }

    constexpr int NIT = KD / 32;   // 32 iterations
    for (int it = 0; it < NIT; ++it) {
        const int cur = it & 1;
        if (wave == 0) __builtin_amdgcn_s_wait_tensorcnt(0);
        __syncthreads();   // tile `it` visible; everyone done reading buf from it-2
        if (wave == 0 && it + 1 < NIT) {
            const unsigned long long off = (unsigned long long)(it + 1) * 64; // 32 elem * 2B
            tdm_load_tile((unsigned)(uintptr_t)&lA[cur ^ 1][0][0], gaA + off);
            tdm_load_tile((unsigned)(uintptr_t)&lB[cur ^ 1][0][0], gaB + off);
        }
        bf16x16 af = load_a_frag(&lA[cur][wave * 16][0], 32);
#pragma unroll
        for (int j = 0; j < 8; ++j) {
            bf16x16 bfr = load_bt_frag(&lB[cur][j * 16][0], 32);
            acc[j] = wmma_bf16(af, bfr, acc[j]);
        }
        __syncthreads();   // reads of buf `cur` done before it+2 overwrites it
    }

    // --- epilogue ---
#pragma unroll
    for (int j = 0; j < 8; ++j) {
#pragma unroll
        for (int r = 0; r < 8; ++r) {
            int m = m0 + wave * 16 + r + 8 * hv;   // C/D layout: M=r+8*(l/16)
            int n = n0 + j * 16 + ln;              //            N=l%16
            float val = acc[j][r] * alpha;
            if (STORE == 0) {
                ((float*)out)[(size_t)m * ND + n] = val;
            } else {
                int b = m >> 11, t = m & (TSEQ - 1);
                int h = n >> 6, d = n & (DHEAD - 1);
                size_t bh = (size_t)b * NHEAD + h;
                size_t idx = (STORE == 1)
                    ? (bh * TSEQ + t) * DHEAD + d
                    : (bh * DHEAD + d) * TSEQ + t;
                ((bf16*)out)[idx] = f2bf(val);
            }
        }
    }
}

// ======================================================================
// Flash attention (causal). q:[bh][t][64] (pre-scaled by 0.125*log2e),
// k:[bh][t][64], v:[bh][d][t] (transposed), y out bf16 [b][t][h*64+d].
// ======================================================================
__global__ __launch_bounds__(256) void attn_kernel(
    const bf16* __restrict__ qm, const bf16* __restrict__ km,
    const bf16* __restrict__ vm, bf16* __restrict__ y)
{
    __shared__ bf16 pb[8][16][32];        // per-wave P bounce buffer

    const int tid  = threadIdx.x;
    const int wave = tid >> 5;
    const int lane = tid & 31;
    const int hv   = lane >> 4;
    const int ln   = lane & 15;
    const int bh   = blockIdx.y;
    const int b    = bh >> 4;
    const int h    = bh & (NHEAD - 1);
    const int qrow0 = blockIdx.x * 128 + wave * 16;

    const bf16* qh = qm + ((size_t)bh * TSEQ + qrow0) * DHEAD;
    const bf16* kh = km + (size_t)bh * TSEQ * DHEAD;
    const bf16* vh = vm + (size_t)bh * DHEAD * TSEQ;

    bf16x16 aq0 = load_a_frag(qh, DHEAD);        // K-chunk 0..31
    bf16x16 aq1 = load_a_frag(qh + 32, DHEAD);   // K-chunk 32..63

    v8f o[4];
    float rm[8], rl[8];
#pragma unroll
    for (int j = 0; j < 4; ++j) o[j] = (v8f)(0.0f);
#pragma unroll
    for (int r = 0; r < 8; ++r) { rm[r] = -1e30f; rl[r] = 0.0f; }

    for (int s0 = 0; s0 < qrow0 + 16; s0 += 32) {
        // ---- scores: two 16x16 tiles over 32 keys, K=64 via 2 WMMAs ----
        const bf16* kbase = kh + (size_t)s0 * DHEAD;
        v8f c0 = (v8f)(0.0f), c1 = (v8f)(0.0f);
        c0 = wmma_bf16(aq0, load_bt_frag(kbase, DHEAD), c0);
        c0 = wmma_bf16(aq1, load_bt_frag(kbase + 32, DHEAD), c0);
        c1 = wmma_bf16(aq0, load_bt_frag(kbase + 16 * DHEAD, DHEAD), c1);
        c1 = wmma_bf16(aq1, load_bt_frag(kbase + 16 * DHEAD + 32, DHEAD), c1);

        // ---- causal mask on diagonal chunks ----
        if (s0 + 31 > qrow0) {
#pragma unroll
            for (int r = 0; r < 8; ++r) {
                int row = qrow0 + r + 8 * hv;
                if (s0 + ln > row)       c0[r] = -1e30f;
                if (s0 + 16 + ln > row)  c1[r] = -1e30f;
            }
        }

        // ---- online softmax (exp2 domain; q pre-scaled by log2e/8) ----
#pragma unroll
        for (int r = 0; r < 8; ++r) {
            float mx = fmaxf(c0[r], c1[r]);
            mx = fmaxf(mx, __shfl_xor(mx, 1, 32));
            mx = fmaxf(mx, __shfl_xor(mx, 2, 32));
            mx = fmaxf(mx, __shfl_xor(mx, 4, 32));
            mx = fmaxf(mx, __shfl_xor(mx, 8, 32));
            float mn   = fmaxf(rm[r], mx);
            float corr = exp2f(rm[r] - mn);
            rm[r] = mn;
            float p0 = exp2f(c0[r] - mn);
            float p1 = exp2f(c1[r] - mn);
            float s  = p0 + p1;
            s += __shfl_xor(s, 1, 32);
            s += __shfl_xor(s, 2, 32);
            s += __shfl_xor(s, 4, 32);
            s += __shfl_xor(s, 8, 32);
            rl[r] = rl[r] * corr + s;
            o[0][r] *= corr; o[1][r] *= corr; o[2][r] *= corr; o[3][r] *= corr;
            pb[wave][r + 8 * hv][ln]      = f2bf(p0);   // D layout -> [m][s]
            pb[wave][r + 8 * hv][ln + 16] = f2bf(p1);
        }

        // wave-local LDS bounce: DS ops are in-order per wave; force wait
        __builtin_amdgcn_wave_barrier();
        asm volatile("s_wait_dscnt 0" ::: "memory");
        bf16x16 pa = load_a_frag(&pb[wave][0][0], 32);
        __builtin_amdgcn_wave_barrier();

        // ---- o += P (16x32) * V (32x64): v stored [d][t] -> contiguous ----
#pragma unroll
        for (int jt = 0; jt < 4; ++jt) {
            bf16x16 bv = load_bt_frag(vh + (size_t)(16 * jt) * TSEQ + s0, TSEQ);
            o[jt] = wmma_bf16(pa, bv, o[jt]);
        }
    }

    // ---- epilogue: y[b][t][h*64+d] = bf16(o / rowsum) ----
#pragma unroll
    for (int jt = 0; jt < 4; ++jt) {
#pragma unroll
        for (int r = 0; r < 8; ++r) {
            int t   = qrow0 + r + 8 * hv;
            int col = h * DHEAD + 16 * jt + ln;
            y[((size_t)b * TSEQ + t) * CDIM + col] = f2bf(o[jt][r] / rl[r]);
        }
    }
}

// ======================================================================
// Pre-pass conversions (bandwidth-trivial at 23.3 TB/s)
// ======================================================================
__global__ __launch_bounds__(256) void cvt_f32_bf16(
    const float* __restrict__ src, bf16* __restrict__ dst, int n4)
{
    int i = blockIdx.x * 256 + threadIdx.x;
    if (i >= n4) return;
    float4 f = *(const float4*)(src + (size_t)i * 4);
    bf16x4 bv = { f2bf(f.x), f2bf(f.y), f2bf(f.z), f2bf(f.w) };
    *(bf16x4*)(dst + (size_t)i * 4) = bv;
}

// w_o (f32 [k][n]) -> bf16 [n][k]   (coalesced writes, 1Mi elements)
__global__ __launch_bounds__(256) void cvt_transpose_wo(
    const float* __restrict__ src, bf16* __restrict__ dst)
{
    int i = blockIdx.x * 256 + threadIdx.x;   // i = n*1024 + k
    int n = i >> 10, k = i & 1023;
    dst[i] = f2bf(src[(size_t)k * CDIM + n]);
}

// ======================================================================
extern "C" void kernel_launch(void* const* d_in, const int* in_sizes, int n_in,
                              void* d_out, int out_size, void* d_ws, size_t ws_size,
                              hipStream_t stream) {
    (void)in_sizes; (void)n_in; (void)out_size; (void)ws_size;
    const float* x  = (const float*)d_in[0];
    const float* wq = (const float*)d_in[1];
    const float* wk = (const float*)d_in[2];
    const float* wv = (const float*)d_in[3];
    const float* wo = (const float*)d_in[4];

    const size_t MC = (size_t)MROWS * CDIM;   // 8Mi
    const size_t WC = (size_t)CDIM * CDIM;    // 1Mi
    bf16* xbf = (bf16*)d_ws;          // 8Mi
    bf16* wqb = xbf + MC;             // 1Mi
    bf16* wkb = wqb + WC;
    bf16* wvb = wkb + WC;
    bf16* wob = wvb + WC;             // transposed [n][k]
    bf16* qws = wob + WC;             // 8Mi each
    bf16* kws = qws + MC;
    bf16* vws = kws + MC;
    bf16* ybf = vws + MC;

    dim3 blk(256);
    const float aq = 0.125f * 1.4426950408889634f;  // 1/sqrt(64) * log2(e)

    cvt_f32_bf16<<<dim3((unsigned)(MC / 4 / 256)), blk, 0, stream>>>(x, xbf, (int)(MC / 4));
    cvt_f32_bf16<<<dim3((unsigned)(WC / 4 / 256)), blk, 0, stream>>>(wq, wqb, (int)(WC / 4));
    cvt_f32_bf16<<<dim3((unsigned)(WC / 4 / 256)), blk, 0, stream>>>(wk, wkb, (int)(WC / 4));
    cvt_f32_bf16<<<dim3((unsigned)(WC / 4 / 256)), blk, 0, stream>>>(wv, wvb, (int)(WC / 4));
    cvt_transpose_wo<<<dim3((unsigned)(WC / 256)), blk, 0, stream>>>(wo, wob);

    dim3 gg(CDIM / 128, MROWS / 128);            // (8, 64)
    gemm_tdm_wmma<1><<<gg, blk, 0, stream>>>(xbf, wqb, (void*)qws, aq);
    gemm_tdm_wmma<1><<<gg, blk, 0, stream>>>(xbf, wkb, (void*)kws, 1.0f);
    gemm_tdm_wmma<2><<<gg, blk, 0, stream>>>(xbf, wvb, (void*)vws, 1.0f);

    attn_kernel<<<dim3(TSEQ / 128, BSZ * NHEAD), blk, 0, stream>>>(qws, kws, vws, ybf);

    gemm_tdm_wmma<0><<<gg, blk, 0, stream>>>(ybf, wob, d_out, 1.0f);
}
